// ScaledDotAttention_50173807952573
// MI455X (gfx1250) — matrix operand order repaired and checked by reading, not compiled, not run on hardware
//
#include <hip/hip_runtime.h>

// ---------------------------------------------------------------------------
// ScaledDotAttention for MI455X (gfx1250): f16 WMMA pipeline, f32 accumulate.
// All GEMM fragments are contiguous vector loads; transposes are materialized
// once (v at projection time, attn via an LDS-tiled transpose) so every WMMA
// kernel runs the same load_b128 -> v_wmma schedule.
// ---------------------------------------------------------------------------

typedef __attribute__((ext_vector_type(16))) _Float16 v16h;
typedef __attribute__((ext_vector_type(8)))  _Float16 h8;
typedef __attribute__((ext_vector_type(4)))  _Float16 h4;
typedef __attribute__((ext_vector_type(2)))  _Float16 h2;
typedef __attribute__((ext_vector_type(2)))  __fp16   fp16x2;
typedef __attribute__((ext_vector_type(8)))  float    v8f;

#define HID 1024
#define SEQ 2048
#define QLEN 2048
#define BATCH 8

__device__ __forceinline__ h2 cvt2(float a, float b) {
    fp16x2 r = __builtin_amdgcn_cvt_pkrtz(a, b);
    return __builtin_bit_cast(h2, r);
}

// ---------------------------------------------------------------------------
// Elementwise f32 -> f16 convert (for the weight matrices). 4 elems/thread.
// ---------------------------------------------------------------------------
__global__ void __launch_bounds__(256)
cvt_f32_f16_kernel(const float* __restrict__ in, _Float16* __restrict__ out) {
    size_t i = ((size_t)blockIdx.x * 256 + threadIdx.x) * 4;
    float4 f = *(const float4*)(in + i);
    h2 a = cvt2(f.x, f.y);
    h2 b = cvt2(f.z, f.w);
    h4 o;
    o[0] = a[0]; o[1] = a[1]; o[2] = b[0]; o[3] = b[1];
    *(h4*)(out + i) = o;
}

// ---------------------------------------------------------------------------
// Projection: Y[m,n] = sum_k X[m,k] * W[n,k] + bias[n]   (nn.Linear)
// X: f32 [M,HID] row-major.  Wh: f16 [HID,HID] row-major (over k -> B frag is
// contiguous).  Y: f16 [M,HID].  One wave = 16x64 output tile, K-loop step 32.
// ---------------------------------------------------------------------------
__global__ void __launch_bounds__(256)
proj_kernel(const float* __restrict__ X, const _Float16* __restrict__ Wh,
            const float* __restrict__ bias, _Float16* __restrict__ Y, int M) {
    const int lane = threadIdx.x & 31;
    const int w    = blockIdx.x * 8 + (threadIdx.x >> 5);
    const int tilesN = HID / 64;            // 16
    const int tn = w % tilesN;
    const int tm = w / tilesN;
    if (tm * 16 >= M) return;               // wave-uniform

    const int row  = lane & 15;
    const int kb   = (lane >> 4) * 8;       // A frag K sub-base
    const int col  = lane & 15;
    const int koff = (lane >> 4) * 8;       // B frag K sub-base

    const float* A0 = X + (size_t)tm * 16 * HID + (size_t)row * HID;

    v8f acc[4] = {};
    for (int k0 = 0; k0 < HID; k0 += 32) {
        const float4* p0 = (const float4*)(A0 + k0 + kb);
        const float4* p1 = (const float4*)(A0 + k0 + 16 + kb);
        float4 f0 = p0[0], f1 = p0[1], f2 = p1[0], f3 = p1[1];
        union { v16h v; h2 p[8]; } ua;
        ua.p[0] = cvt2(f0.x, f0.y); ua.p[1] = cvt2(f0.z, f0.w);
        ua.p[2] = cvt2(f1.x, f1.y); ua.p[3] = cvt2(f1.z, f1.w);
        ua.p[4] = cvt2(f2.x, f2.y); ua.p[5] = cvt2(f2.z, f2.w);
        ua.p[6] = cvt2(f3.x, f3.y); ua.p[7] = cvt2(f3.z, f3.w);
#pragma unroll
        for (int t = 0; t < 4; ++t) {
            const _Float16* pb =
                Wh + (size_t)(tn * 64 + t * 16 + col) * HID + k0 + koff;
            v16h bfrag = __builtin_shufflevector(*(const h8*)pb, *(const h8*)(pb + 16), 0,1,2,3,4,5,6,7,8,9,10,11,12,13,14,15);
            acc[t] = __builtin_amdgcn_wmma_f32_16x16x32_f16(
                false, ua.v, false, bfrag, (short)0, acc[t], false, false);
        }
    }

    const int mrow = tm * 16 + 8 * (lane >> 4);
#pragma unroll
    for (int t = 0; t < 4; ++t) {
        const int n = tn * 64 + t * 16 + col;
        const float bv = bias[n];
#pragma unroll
        for (int r = 0; r < 8; ++r)
            Y[(size_t)(mrow + r) * HID + n] = (_Float16)(acc[t][r] + bv);
    }
}

// ---------------------------------------------------------------------------
// Projection with transposed store (for v):  Yt[b, n, s] = proj(v)[b, s, n]
// Same GEMM as proj_kernel; output goes to [BATCH][HID][SEQ] so the context
// kernel's B fragments become contiguous over s. One-time uncoalesced stores.
// ---------------------------------------------------------------------------
__global__ void __launch_bounds__(256)
proj_t_kernel(const float* __restrict__ X, const _Float16* __restrict__ Wh,
              const float* __restrict__ bias, _Float16* __restrict__ Yt, int M) {
    const int lane = threadIdx.x & 31;
    const int w    = blockIdx.x * 8 + (threadIdx.x >> 5);
    const int tilesN = HID / 64;            // 16
    const int tn = w % tilesN;
    const int tm = w / tilesN;
    if (tm * 16 >= M) return;

    const int row  = lane & 15;
    const int kb   = (lane >> 4) * 8;
    const int col  = lane & 15;
    const int koff = (lane >> 4) * 8;

    const float* A0 = X + (size_t)tm * 16 * HID + (size_t)row * HID;

    v8f acc[4] = {};
    for (int k0 = 0; k0 < HID; k0 += 32) {
        const float4* p0 = (const float4*)(A0 + k0 + kb);
        const float4* p1 = (const float4*)(A0 + k0 + 16 + kb);
        float4 f0 = p0[0], f1 = p0[1], f2 = p1[0], f3 = p1[1];
        union { v16h v; h2 p[8]; } ua;
        ua.p[0] = cvt2(f0.x, f0.y); ua.p[1] = cvt2(f0.z, f0.w);
        ua.p[2] = cvt2(f1.x, f1.y); ua.p[3] = cvt2(f1.z, f1.w);
        ua.p[4] = cvt2(f2.x, f2.y); ua.p[5] = cvt2(f2.z, f2.w);
        ua.p[6] = cvt2(f3.x, f3.y); ua.p[7] = cvt2(f3.z, f3.w);
#pragma unroll
        for (int t = 0; t < 4; ++t) {
            const _Float16* pb =
                Wh + (size_t)(tn * 64 + t * 16 + col) * HID + k0 + koff;
            v16h bfrag = __builtin_shufflevector(*(const h8*)pb, *(const h8*)(pb + 16), 0,1,2,3,4,5,6,7,8,9,10,11,12,13,14,15);
            acc[t] = __builtin_amdgcn_wmma_f32_16x16x32_f16(
                false, ua.v, false, bfrag, (short)0, acc[t], false, false);
        }
    }

    // Transposed store: tile rows are s (within batch bb), cols are n.
    const int mrow = tm * 16 + 8 * (lane >> 4);
    const int bb   = (tm * 16) / SEQ;        // 16-row tile never crosses batch
    const int srow = mrow - bb * SEQ;
    _Float16* Ot = Yt + (size_t)bb * HID * SEQ;
#pragma unroll
    for (int t = 0; t < 4; ++t) {
        const int n = tn * 64 + t * 16 + col;
        const float bv = bias[n];
#pragma unroll
        for (int r = 0; r < 8; ++r)
            Ot[(size_t)n * SEQ + srow + r] = (_Float16)(acc[t][r] + bv);
    }
}

// ---------------------------------------------------------------------------
// Scores: attn[b,s,q] = (1/32) * sum_h kh[b,s,h] * qh[b,q,h]
// Both operands f16, B fragment contiguous over h (qh acts like W above).
// ---------------------------------------------------------------------------
__global__ void __launch_bounds__(256)
scores_kernel(const _Float16* __restrict__ kh, const _Float16* __restrict__ qh,
              float* __restrict__ attn) {
    const float scale = 0.03125f;           // 1/sqrt(1024)
    const int lane = threadIdx.x & 31;
    const int w    = blockIdx.x * 8 + (threadIdx.x >> 5);
    const int tilesN = QLEN / 64;           // 32
    const int tilesM = SEQ / 16;            // 128
    const int perB = tilesM * tilesN;       // 4096
    const int b  = w / perB;
    const int r2 = w % perB;
    const int tm = r2 / tilesN;
    const int tn = r2 % tilesN;

    const int row = lane & 15, kb = (lane >> 4) * 8;
    const int col = lane & 15, koff = (lane >> 4) * 8;

    const _Float16* A0 = kh + (size_t)b * SEQ * HID +
                         (size_t)(tm * 16 + row) * HID;
    const _Float16* B0 = qh + (size_t)b * QLEN * HID;

    v8f acc[4] = {};
    for (int k0 = 0; k0 < HID; k0 += 32) {
        union { v16h v; h8 o[2]; } ua;
        ua.o[0] = *(const h8*)(A0 + k0 + kb);
        ua.o[1] = *(const h8*)(A0 + k0 + 16 + kb);
#pragma unroll
        for (int t = 0; t < 4; ++t) {
            const _Float16* pb =
                B0 + (size_t)(tn * 64 + t * 16 + col) * HID + k0 + koff;
            v16h bfrag = __builtin_shufflevector(*(const h8*)pb, *(const h8*)(pb + 16), 0,1,2,3,4,5,6,7,8,9,10,11,12,13,14,15);
            acc[t] = __builtin_amdgcn_wmma_f32_16x16x32_f16(
                false, ua.v, false, bfrag, (short)0, acc[t], false, false);
        }
    }

    float* O = attn + (size_t)b * SEQ * QLEN;
    const int srow = tm * 16 + 8 * (lane >> 4);
#pragma unroll
    for (int t = 0; t < 4; ++t) {
        const int q = tn * 64 + t * 16 + col;
#pragma unroll
        for (int r = 0; r < 8; ++r)
            O[(size_t)(srow + r) * QLEN + q] = acc[t][r] * scale;
    }
}

// ---------------------------------------------------------------------------
// Column softmax over the S axis of attn[b,S,Q], in place. One thread per
// (b,q) column; consecutive threads hit consecutive columns -> coalesced rows.
// Online max+sum pass, then normalize pass.
// ---------------------------------------------------------------------------
__global__ void __launch_bounds__(256)
softmax_kernel(float* __restrict__ attn) {
    const int idx = blockIdx.x * 256 + threadIdx.x;   // B*Q threads
    const int b = idx / QLEN;
    const int q = idx % QLEN;
    float* col = attn + (size_t)b * SEQ * QLEN + q;

    float m = -3.4e38f, sum = 0.f;
    for (int s = 0; s < SEQ; ++s) {
        float x  = col[(size_t)s * QLEN];
        float nm = fmaxf(m, x);
        sum = sum * __expf(m - nm) + __expf(x - nm);
        m = nm;
    }
    const float inv = 1.0f / sum;
    for (int s = 0; s < SEQ; ++s) {
        float x = col[(size_t)s * QLEN];
        col[(size_t)s * QLEN] = __expf(x - m) * inv;
    }
}

// ---------------------------------------------------------------------------
// Tiled transpose + f32->f16: attnT[b,q,s] = (f16) attn[b,s,q]
// 32x32 tiles through LDS (padded stride 33), fully coalesced both sides.
// blockDim = (32,8); grid = (QLEN/32, SEQ/32, BATCH).
// ---------------------------------------------------------------------------
__global__ void __launch_bounds__(256)
transpose_attn_kernel(const float* __restrict__ attn,
                      _Float16* __restrict__ attnT) {
    __shared__ _Float16 tile[32][33];
    const int b  = blockIdx.z;
    const int q0 = blockIdx.x * 32;
    const int s0 = blockIdx.y * 32;
    const int tx = threadIdx.x;             // 0..31
    const int ty = threadIdx.y;             // 0..7

    const float* in = attn + (size_t)b * SEQ * QLEN;
#pragma unroll
    for (int k = 0; k < 4; ++k) {
        const int s = s0 + ty + 8 * k;
        tile[tx][ty + 8 * k] = (_Float16)in[(size_t)s * QLEN + q0 + tx];
    }
    __syncthreads();
    _Float16* out = attnT + (size_t)b * QLEN * SEQ;
#pragma unroll
    for (int k = 0; k < 4; ++k) {
        const int q = q0 + ty + 8 * k;
        out[(size_t)q * SEQ + s0 + tx] = tile[ty + 8 * k][tx];
    }
}

// ---------------------------------------------------------------------------
// Context: ctx[b,q,h] = sum_s attnT[b,q,s] * vhT[b,h,s]
// A row-major over s (contiguous h8 loads); B fragment built from two h8 loads
// from vhT. Same schedule as scores_kernel; K = SEQ (64 steps).
// ---------------------------------------------------------------------------
__global__ void __launch_bounds__(256)
context_kernel(const _Float16* __restrict__ attnT,
               const _Float16* __restrict__ vhT, float* __restrict__ ctx) {
    const int lane = threadIdx.x & 31;
    const int w    = blockIdx.x * 8 + (threadIdx.x >> 5);
    const int tilesN = HID / 64;            // 16
    const int tilesM = QLEN / 16;           // 128
    const int perB = tilesM * tilesN;       // 2048
    const int b  = w / perB;
    const int r2 = w % perB;
    const int tm = r2 / tilesN;
    const int tn = r2 % tilesN;

    const int row = lane & 15, kb = (lane >> 4) * 8;
    const int col = lane & 15, koff = (lane >> 4) * 8;

    const _Float16* A0 = attnT + (size_t)b * QLEN * SEQ +
                         (size_t)(tm * 16 + row) * SEQ;
    const _Float16* B0 = vhT + (size_t)b * HID * SEQ;

    v8f acc[4] = {};
    for (int k0 = 0; k0 < SEQ; k0 += 32) {
        union { v16h v; h8 o[2]; } ua;
        ua.o[0] = *(const h8*)(A0 + k0 + kb);
        ua.o[1] = *(const h8*)(A0 + k0 + 16 + kb);
#pragma unroll
        for (int t = 0; t < 4; ++t) {
            const _Float16* pb =
                B0 + (size_t)(tn * 64 + t * 16 + col) * SEQ + k0 + koff;
            v16h bfrag = __builtin_shufflevector(*(const h8*)pb, *(const h8*)(pb + 16), 0,1,2,3,4,5,6,7,8,9,10,11,12,13,14,15);
            acc[t] = __builtin_amdgcn_wmma_f32_16x16x32_f16(
                false, ua.v, false, bfrag, (short)0, acc[t], false, false);
        }
    }

    float* O = ctx + (size_t)b * QLEN * HID;
    const int qrow = tm * 16 + 8 * (lane >> 4);
#pragma unroll
    for (int t = 0; t < 4; ++t) {
        const int h = tn * 64 + t * 16 + col;
#pragma unroll
        for (int r = 0; r < 8; ++r)
            O[(size_t)(qrow + r) * HID + h] = acc[t][r];
    }
}

// ---------------------------------------------------------------------------
extern "C" void kernel_launch(void* const* d_in, const int* in_sizes, int n_in,
                              void* d_out, int out_size, void* d_ws, size_t ws_size,
                              hipStream_t stream) {
    const float* queries = (const float*)d_in[0];
    const float* keys    = (const float*)d_in[1];
    const float* values  = (const float*)d_in[2];
    const float* Wq = (const float*)d_in[3];
    const float* bq = (const float*)d_in[4];
    const float* Wk = (const float*)d_in[5];
    const float* bk = (const float*)d_in[6];
    const float* Wv = (const float*)d_in[7];
    const float* bv = (const float*)d_in[8];

    float* ctx_out  = (float*)d_out;                        // [B,Q,H]
    float* attn_out = ctx_out + (size_t)BATCH * QLEN * HID; // [B,S,Q]

    // Workspace (f16): Wq/Wk/Wv converted, qh, kh, vhT, attnT  (~174 MB)
    _Float16* wqh   = (_Float16*)d_ws;
    _Float16* wkh   = wqh + (size_t)HID * HID;
    _Float16* wvh   = wkh + (size_t)HID * HID;
    _Float16* qh    = wvh + (size_t)HID * HID;
    _Float16* kh    = qh + (size_t)BATCH * QLEN * HID;
    _Float16* vhT   = kh + (size_t)BATCH * SEQ * HID;       // [B][HID][SEQ]
    _Float16* attnT = vhT + (size_t)BATCH * HID * SEQ;      // [B][QLEN][SEQ]

    const int cvtBlocks = (HID * HID) / 1024;               // 4 elems/thread
    cvt_f32_f16_kernel<<<cvtBlocks, 256, 0, stream>>>(Wq, wqh);
    cvt_f32_f16_kernel<<<cvtBlocks, 256, 0, stream>>>(Wk, wkh);
    cvt_f32_f16_kernel<<<cvtBlocks, 256, 0, stream>>>(Wv, wvh);

    // Projections: M=16384, 16x64 tile per wave -> 2048 blocks of 8 waves
    proj_kernel<<<2048, 256, 0, stream>>>(queries, wqh, bq, qh, BATCH * QLEN);
    proj_kernel<<<2048, 256, 0, stream>>>(keys,    wkh, bk, kh, BATCH * SEQ);
    proj_t_kernel<<<2048, 256, 0, stream>>>(values, wvh, bv, vhT, BATCH * SEQ);

    // Scores: 8 * 128 * 32 = 32768 waves -> 4096 blocks
    scores_kernel<<<4096, 256, 0, stream>>>(kh, qh, attn_out);

    // Softmax: B*Q = 16384 columns -> 64 blocks
    softmax_kernel<<<64, 256, 0, stream>>>(attn_out);

    // Transpose attn (f32->f16): 64x64x8 tiles of 32x32
    transpose_attn_kernel<<<dim3(QLEN / 32, SEQ / 32, BATCH), dim3(32, 8), 0,
                            stream>>>(attn_out, attnT);

    // Context: 8 * 128 * 16 = 16384 waves -> 2048 blocks
    context_kernel<<<2048, 256, 0, stream>>>(attnT, vhT, ctx_out);
}
